// SlimeMoldEncoder_51445118271636
// MI455X (gfx1250) — matrix operand order repaired and checked by reading, not compiled, not run on hardware
//
#include <hip/hip_runtime.h>
#include <hip/hip_bf16.h>
#include <math.h>

// ---------------------------------------------------------------------------
// SlimeMoldEncoder for MI455X (gfx1250, wave32, WMMA)
// Sequential scan: per-step f16 WMMA GEMMs. Weights live in L2 as f16;
// each workgroup async-stages a 64x64 weight tile into LDS
// (global_load_async_to_lds_b128, double-buffered) shared by 8 waves.
// ---------------------------------------------------------------------------

typedef __attribute__((ext_vector_type(16))) _Float16 v16h;
typedef __attribute__((ext_vector_type(8)))  _Float16 v8h;
typedef __attribute__((ext_vector_type(8)))  float    v8f;

#define DMODEL 1024
#define PHD    4096   // P * HD = 64 * 64
#define BATCH  128
#define SEQ    1024
#define BD     (BATCH * DMODEL)   // 131072
#define NPOD   64
#define HDIM   64

// LDS B-tile: 64 columns x 64 K-values (f16). Row stride padded to 144B
// (72 f16 = 36 dwords): 36*16 = 9*64, so all 16 lanes of a half-wave start
// on distinct banks -> conflict-free ds_load_b128.
#define KCH    64                 // K-chunk staged per double-buffer step
#define BROW   72                 // f16 elements per padded LDS row
#define BTILE  (64 * BROW)        // one buffer (9216 f16 = 18KB total x2)

__device__ __forceinline__ void async_b128(_Float16* lds_dst, const _Float16* gsrc) {
    uint32_t loff = (uint32_t)(uintptr_t)lds_dst;
    asm volatile("global_load_async_to_lds_b128 %0, %1, off"
                 :: "v"(loff), "v"(gsrc) : "memory");
}
__device__ __forceinline__ void wait_async0() {
    asm volatile("s_wait_asynccnt 0x0" ::: "memory");
}

__device__ __forceinline__ v16h cat8(v8h lo, v8h hi) {
    return __builtin_shufflevector(lo, hi, 0,1,2,3,4,5,6,7,8,9,10,11,12,13,14,15);
}

// ---------------------------------------------------------------------------
// f16 WMMA GEMM:  C[128,N] = A[128,K] (f16 row-major) * Bt[N,K]^T (f16)
// Block = 256 threads = 8 waves; block owns a 128x64 strip (grid.x = N/64).
// Wave w handles M-tile w; the 64x64 weight tile is async-staged to LDS
// once per K-chunk and shared by all 8 waves (8x less L2 weight traffic).
// EPI 0: store f32 -> Cf | 1: tanh(c+bias)->Ch f16 | 2: c+bias -> Cf
// ---------------------------------------------------------------------------
template <int EPI>
__global__ __launch_bounds__(256)
void gemm_wmma_f16(const _Float16* __restrict__ A,
                   const _Float16* __restrict__ Bt,
                   const float*    __restrict__ bias,
                   float*          __restrict__ Cf,
                   _Float16*       __restrict__ Ch,
                   int N, int K) {
    __shared__ _Float16 ldsb[2][BTILE];

    const int tid  = threadIdx.x;
    const int lane = tid & 31;
    const int wave = tid >> 5;               // = M-tile index (M = 128)
    const int n0   = blockIdx.x << 6;        // base column of 64-wide strip

    const int h = lane >> 4;                 // lane half (0/1)
    const int l = lane & 15;
    const int m = (wave << 4) + l;           // A row this lane feeds

    // staging map: thread -> two 16B chunks of the 64x64 f16 tile
    const int scol = tid >> 2;               // 0..63 (column within strip)
    const int sseg = tid & 3;                // 0..3  (8-f16 segment, + segment 4..7)
    const _Float16* gB   = Bt + (size_t)(n0 + scol) * K + sseg * 8;
    _Float16* lB0        = &ldsb[0][scol * BROW + sseg * 8];
    _Float16* lB1        = &ldsb[1][scol * BROW + sseg * 8];

    const _Float16* __restrict__ arow = A + (size_t)m * K;

    v8f acc0 = {}, acc1 = {}, acc2 = {}, acc3 = {};

    // prologue: stage first K-chunk into buffer 0 (two 16B pieces per thread)
    async_b128(lB0,      gB);
    async_b128(lB0 + 32, gB + 32);

    int ib = 0;
    for (int kk = 0; kk < K; kk += KCH, ib ^= 1) {
        // A fragments 16x32 x2 (per-lane contiguous 16B chunks, global)
        v16h a0 = cat8(*(const v8h*)(arow + kk + h * 16),
                       *(const v8h*)(arow + kk + h * 16 + 8));
        v16h a1 = cat8(*(const v8h*)(arow + kk + 32 + h * 16),
                       *(const v8h*)(arow + kk + 32 + h * 16 + 8));

        // current buffer staged & previous reads of the other buffer done
        wait_async0();
        __syncthreads();

        // kick off async staging of the next K-chunk into the other buffer
        if (kk + KCH < K) {
            _Float16* nb = ib ? lB0 : lB1;
            async_b128(nb,      gB + kk + KCH);
            async_b128(nb + 32, gB + kk + KCH + 32);
        }

        // B fragments from LDS: sub-tile s covers columns n0+s*16..+15,
        // K sub-chunk c covers kk+c*32..+31; lane holds K = +h*16..+15.
        const _Float16* lb = &ldsb[ib][l * BROW + h * 16];
        v16h b00 = cat8(*(const v8h*)(lb),                 *(const v8h*)(lb + 8));
        v16h b10 = cat8(*(const v8h*)(lb + 16 * BROW),     *(const v8h*)(lb + 16 * BROW + 8));
        v16h b20 = cat8(*(const v8h*)(lb + 32 * BROW),     *(const v8h*)(lb + 32 * BROW + 8));
        v16h b30 = cat8(*(const v8h*)(lb + 48 * BROW),     *(const v8h*)(lb + 48 * BROW + 8));
        v16h b01 = cat8(*(const v8h*)(lb + 32),            *(const v8h*)(lb + 40));
        v16h b11 = cat8(*(const v8h*)(lb + 16 * BROW + 32),*(const v8h*)(lb + 16 * BROW + 40));
        v16h b21 = cat8(*(const v8h*)(lb + 32 * BROW + 32),*(const v8h*)(lb + 32 * BROW + 40));
        v16h b31 = cat8(*(const v8h*)(lb + 48 * BROW + 32),*(const v8h*)(lb + 48 * BROW + 40));

        // keep the 16 ds_load_b128 grouped ahead of the 8 WMMAs
        __builtin_amdgcn_sched_barrier(0);

        acc0 = __builtin_amdgcn_wmma_f32_16x16x32_f16(false, a0, false, b00,
                                                      (short)0, acc0, false, false);
        acc1 = __builtin_amdgcn_wmma_f32_16x16x32_f16(false, a0, false, b10,
                                                      (short)0, acc1, false, false);
        acc2 = __builtin_amdgcn_wmma_f32_16x16x32_f16(false, a0, false, b20,
                                                      (short)0, acc2, false, false);
        acc3 = __builtin_amdgcn_wmma_f32_16x16x32_f16(false, a0, false, b30,
                                                      (short)0, acc3, false, false);
        acc0 = __builtin_amdgcn_wmma_f32_16x16x32_f16(false, a1, false, b01,
                                                      (short)0, acc0, false, false);
        acc1 = __builtin_amdgcn_wmma_f32_16x16x32_f16(false, a1, false, b11,
                                                      (short)0, acc1, false, false);
        acc2 = __builtin_amdgcn_wmma_f32_16x16x32_f16(false, a1, false, b21,
                                                      (short)0, acc2, false, false);
        acc3 = __builtin_amdgcn_wmma_f32_16x16x32_f16(false, a1, false, b31,
                                                      (short)0, acc3, false, false);

        // all waves done reading buffer ib before it is restaged next iter
        __syncthreads();
    }

    // C/D layout: vgpr j, lanes 0-15 -> M=j, lanes 16-31 -> M=8+j; N = lane%16
    v8f accs[4] = {acc0, acc1, acc2, acc3};
#pragma unroll
    for (int s = 0; s < 4; ++s) {
        const int n = n0 + s * 16 + l;
#pragma unroll
        for (int j = 0; j < 8; ++j) {
            const int row = (wave << 4) + h * 8 + j;
            float c = accs[s][j];
            if (EPI == 0) {
                Cf[(size_t)row * N + n] = c;
            } else if (EPI == 1) {
                Ch[(size_t)row * N + n] = (_Float16)tanhf(c + bias[n]);
            } else {
                Cf[(size_t)row * N + n] = c + bias[n];
            }
        }
    }
}

// ---------------------------------------------------------------------------
// Weight convert + transpose: W[K,N] f32 -> Wt[N,K] f16
// ---------------------------------------------------------------------------
__global__ void cvt_transpose_kernel(const float* __restrict__ W,
                                     _Float16* __restrict__ Wt,
                                     int K, int N) {
    size_t i = (size_t)blockIdx.x * blockDim.x + threadIdx.x;
    if (i < (size_t)K * N) {
        int k = (int)(i / N);
        int n = (int)(i % N);
        Wt[(size_t)n * K + k] = (_Float16)W[i];
    }
}

// Convert one timestep of src to f16
__global__ void cvt_f16_kernel(const float* __restrict__ x,
                               _Float16* __restrict__ x16, int n) {
    int i = blockIdx.x * blockDim.x + threadIdx.x;
    if (i < n) x16[i] = (_Float16)x[i];
}

// Zero-init organism state (f32 + f16 mirrors)
__global__ void init_state_kernel(float* __restrict__ st,
                                  _Float16* __restrict__ st16, int n) {
    int i = blockIdx.x * blockDim.x + threadIdx.x;
    if (i < n) { st[i] = 0.0f; st16[i] = (_Float16)0.0f; }
}

// ---------------------------------------------------------------------------
// Pseudopod pool attention: per batch row b (one block, 64 threads = P pods)
// ---------------------------------------------------------------------------
__global__ __launch_bounds__(NPOD)
void attn_kernel(const float* __restrict__ q,
                 const float* __restrict__ k,
                 const float* __restrict__ v,
                 _Float16* __restrict__ wv16) {
    const int b = blockIdx.x;
    const int p = threadIdx.x;
    __shared__ float vit[NPOD];

    const size_t base = (size_t)b * PHD + (size_t)p * HDIM;
    const float* qb = q + base;
    const float* kb = k + base;

    float acc = 0.0f;
#pragma unroll 8
    for (int i = 0; i < HDIM; ++i) acc += qb[i] * kb[i];
    vit[p] = acc * 0.125f;                 // 1/sqrt(64)
    __syncthreads();

    float mx = -1e30f;
#pragma unroll 8
    for (int i = 0; i < NPOD; ++i) mx = fmaxf(mx, vit[i]);
    float sum = 0.0f;
#pragma unroll 8
    for (int i = 0; i < NPOD; ++i) sum += __expf(vit[i] - mx);
    const float alloc = __expf(vit[p] - mx) / sum;

    const float* vb = v + base;
    _Float16* ob = wv16 + base;
#pragma unroll 8
    for (int i = 0; i < HDIM; ++i) ob[i] = (_Float16)(alloc * vb[i]);
}

// ---------------------------------------------------------------------------
// Gate update: z = sigmoid(zpre + xu); state = z*state + (1-z)*tanh(flow)
// ---------------------------------------------------------------------------
__global__ void gate_kernel(const float* __restrict__ zpre,
                            const float* __restrict__ xu,
                            const float* __restrict__ flow,
                            float* __restrict__ state,
                            _Float16* __restrict__ state16, int n) {
    int i = blockIdx.x * blockDim.x + threadIdx.x;
    if (i < n) {
        float z  = 1.0f / (1.0f + __expf(-(zpre[i] + xu[i])));
        float ns = z * state[i] + (1.0f - z) * tanhf(flow[i]);
        state[i]   = ns;
        state16[i] = (_Float16)ns;
    }
}

// ---------------------------------------------------------------------------
// Host orchestration
// ---------------------------------------------------------------------------
extern "C" void kernel_launch(void* const* d_in, const int* in_sizes, int n_in,
                              void* d_out, int out_size, void* d_ws, size_t ws_size,
                              hipStream_t stream) {
    (void)in_sizes; (void)n_in; (void)out_size; (void)ws_size;

    const float* src     = (const float*)d_in[0];
    const float* W_sense = (const float*)d_in[1];
    const float* b_sense = (const float*)d_in[2];
    const float* Wq      = (const float*)d_in[3];
    const float* Wk      = (const float*)d_in[4];
    const float* Wv      = (const float*)d_in[5];
    const float* Wo      = (const float*)d_in[6];
    const float* Wz      = (const float*)d_in[7];
    const float* Uz      = (const float*)d_in[8];
    const float* W_out   = (const float*)d_in[9];
    const float* b_out   = (const float*)d_in[10];
    float* out = (float*)d_out;

    // deterministic scratch carve-out
    char* ws = (char*)d_ws;
    size_t off = 0;
    auto carve = [&](size_t bytes) -> void* {
        void* p = ws + off;
        off = (off + bytes + 255) & ~(size_t)255;
        return p;
    };

    _Float16* wt_sense = (_Float16*)carve((size_t)DMODEL * DMODEL * 2);
    _Float16* wt_q     = (_Float16*)carve((size_t)DMODEL * PHD * 2);
    _Float16* wt_k     = (_Float16*)carve((size_t)DMODEL * PHD * 2);
    _Float16* wt_v     = (_Float16*)carve((size_t)DMODEL * PHD * 2);
    _Float16* wt_o     = (_Float16*)carve((size_t)PHD * DMODEL * 2);
    _Float16* wt_z     = (_Float16*)carve((size_t)DMODEL * DMODEL * 2);
    _Float16* wt_u     = (_Float16*)carve((size_t)DMODEL * DMODEL * 2);
    _Float16* wt_out   = (_Float16*)carve((size_t)DMODEL * DMODEL * 2);

    _Float16* x16     = (_Float16*)carve((size_t)BD * 2);
    _Float16* sense16 = (_Float16*)carve((size_t)BD * 2);
    float*    st32    = (float*)   carve((size_t)BD * 4);
    _Float16* st16    = (_Float16*)carve((size_t)BD * 2);
    float*    qbuf    = (float*)   carve((size_t)BATCH * PHD * 4);
    float*    kbuf    = (float*)   carve((size_t)BATCH * PHD * 4);
    float*    vbuf    = (float*)   carve((size_t)BATCH * PHD * 4);
    _Float16* wv16    = (_Float16*)carve((size_t)BATCH * PHD * 2);
    float*    zpre    = (float*)   carve((size_t)BD * 4);
    float*    xu      = (float*)   carve((size_t)BD * 4);
    float*    flow    = (float*)   carve((size_t)BD * 4);

    const int T256 = 256;
    auto cdiv = [](size_t a, size_t b) { return (int)((a + b - 1) / b); };

    // ---- prologue: weights -> f16, pre-transposed to [N,K] ----
    cvt_transpose_kernel<<<cdiv((size_t)DMODEL * DMODEL, T256), T256, 0, stream>>>(W_sense, wt_sense, DMODEL, DMODEL);
    cvt_transpose_kernel<<<cdiv((size_t)DMODEL * PHD,    T256), T256, 0, stream>>>(Wq, wt_q, DMODEL, PHD);
    cvt_transpose_kernel<<<cdiv((size_t)DMODEL * PHD,    T256), T256, 0, stream>>>(Wk, wt_k, DMODEL, PHD);
    cvt_transpose_kernel<<<cdiv((size_t)DMODEL * PHD,    T256), T256, 0, stream>>>(Wv, wt_v, DMODEL, PHD);
    cvt_transpose_kernel<<<cdiv((size_t)PHD * DMODEL,    T256), T256, 0, stream>>>(Wo, wt_o, PHD, DMODEL);
    cvt_transpose_kernel<<<cdiv((size_t)DMODEL * DMODEL, T256), T256, 0, stream>>>(Wz, wt_z, DMODEL, DMODEL);
    cvt_transpose_kernel<<<cdiv((size_t)DMODEL * DMODEL, T256), T256, 0, stream>>>(Uz, wt_u, DMODEL, DMODEL);
    cvt_transpose_kernel<<<cdiv((size_t)DMODEL * DMODEL, T256), T256, 0, stream>>>(W_out, wt_out, DMODEL, DMODEL);

    init_state_kernel<<<cdiv(BD, T256), T256, 0, stream>>>(st32, st16, BD);

    // GEMM grids: one block per 64-wide N strip (block covers all of M=128)
    const dim3 blk(256);
    const dim3 grid_n1k(DMODEL / 64);   // N=1024 -> 16 blocks
    const dim3 grid_n4k(PHD / 64);      // N=4096 -> 64 blocks

    // ---- sequential organism scan ----
    for (int t = 0; t < SEQ; ++t) {
        const float* x_t = src + (size_t)t * BD;
        float* out_t = out + (size_t)t * BD;

        cvt_f16_kernel<<<cdiv(BD, T256), T256, 0, stream>>>(x_t, x16, BD);

        // sense = tanh(x @ W_sense + b_sense)          [128,1024] (f16)
        gemm_wmma_f16<1><<<grid_n1k, blk, 0, stream>>>(x16, wt_sense, b_sense,
                                                       nullptr, sense16,
                                                       DMODEL, DMODEL);
        // k = sense @ Wk ; v = sense @ Wv              [128,4096] (f32)
        gemm_wmma_f16<0><<<grid_n4k, blk, 0, stream>>>(sense16, wt_k, nullptr,
                                                       kbuf, nullptr,
                                                       PHD, DMODEL);
        gemm_wmma_f16<0><<<grid_n4k, blk, 0, stream>>>(sense16, wt_v, nullptr,
                                                       vbuf, nullptr,
                                                       PHD, DMODEL);
        // q = state @ Wq                               [128,4096] (f32)
        gemm_wmma_f16<0><<<grid_n4k, blk, 0, stream>>>(st16, wt_q, nullptr,
                                                       qbuf, nullptr,
                                                       PHD, DMODEL);
        // zpre = state @ Wz ; xu = x @ Uz              [128,1024] (f32)
        gemm_wmma_f16<0><<<grid_n1k, blk, 0, stream>>>(st16, wt_z, nullptr,
                                                       zpre, nullptr,
                                                       DMODEL, DMODEL);
        gemm_wmma_f16<0><<<grid_n1k, blk, 0, stream>>>(x16, wt_u, nullptr,
                                                       xu, nullptr,
                                                       DMODEL, DMODEL);
        // pseudopod pool softmax + weighted v          -> wv16 (f16)
        attn_kernel<<<BATCH, NPOD, 0, stream>>>(qbuf, kbuf, vbuf, wv16);

        // flow = wv @ Wo                               [128,1024] (f32)
        gemm_wmma_f16<0><<<grid_n1k, blk, 0, stream>>>(wv16, wt_o, nullptr,
                                                       flow, nullptr,
                                                       DMODEL, PHD);
        // gated state update
        gate_kernel<<<cdiv(BD, T256), T256, 0, stream>>>(zpre, xu, flow,
                                                         st32, st16, BD);
        // out_t = new_state @ W_out + b_out            [128,1024] (f32)
        gemm_wmma_f16<2><<<grid_n1k, blk, 0, stream>>>(st16, wt_out, b_out,
                                                       out_t, nullptr,
                                                       DMODEL, DMODEL);
    }
}
